// BitFeedForward_28252294873583
// MI455X (gfx1250) — 8-bit matrix operand order repaired and checked by reading, compiled, not run on hardware
//
#include <hip/hip_runtime.h>
#include <math.h>

typedef __attribute__((ext_vector_type(8))) int v8i;

#define FF_EPS 1e-8f

__global__ void absmean_partial(const float* __restrict__ w, size_t n,
                                float* __restrict__ part) {
    __shared__ float red[256];
    float s = 0.f;
    for (size_t i = (size_t)blockIdx.x * blockDim.x + threadIdx.x; i < n;
         i += (size_t)gridDim.x * blockDim.x)
        s += fabsf(w[i]);
    red[threadIdx.x] = s;
    __syncthreads();
    for (int o = 128; o > 0; o >>= 1) {
        if ((int)threadIdx.x < o) red[threadIdx.x] += red[threadIdx.x + o];
        __syncthreads();
    }
    if (threadIdx.x == 0) part[blockIdx.x] = red[0];
}

__global__ void finalize_scales(const float* __restrict__ part,
                                float* __restrict__ scl, float inv_count) {
    __shared__ float red[256];
    for (int w = 0; w < 3; ++w) {
        const float* p = part + w * 1024;
        float s = p[threadIdx.x] + p[threadIdx.x + 256] + p[threadIdx.x + 512] +
                  p[threadIdx.x + 768];
        red[threadIdx.x] = s;
        __syncthreads();
        for (int o = 128; o > 0; o >>= 1) {
            if ((int)threadIdx.x < o) red[threadIdx.x] += red[threadIdx.x + o];
            __syncthreads();
        }
        if (threadIdx.x == 0) {
            float m = red[0] * inv_count;
            scl[4 + w] = m + FF_EPS;
            scl[8 + w] = atanhf(m);
        }
        __syncthreads();
    }
}

__global__ void ternary_quant_w(const float* __restrict__ w,
                                const float* __restrict__ scl, int idx,
                                signed char* __restrict__ wq, size_t n) {
    float inv = 1.0f / scl[4 + idx];
    for (size_t i = (size_t)blockIdx.x * blockDim.x + threadIdx.x; i < n;
         i += (size_t)gridDim.x * blockDim.x)
        wq[i] = (signed char)(int)rintf(tanhf(w[i] * inv));
}

__global__ void rmsnorm_actquant(const float* __restrict__ x, int D,
                                 const float* __restrict__ g1,
                                 signed char* __restrict__ q1,
                                 float* __restrict__ deq1,
                                 const float* __restrict__ g2,
                                 signed char* __restrict__ q2,
                                 float* __restrict__ deq2) {
    __shared__ float red[256];
    const int tid = threadIdx.x;
    const size_t rowoff = (size_t)blockIdx.x * (size_t)D;
    float v[32];
    float ss = 0.f;
#pragma unroll
    for (int i = 0; i < 32; ++i) {
        int c = tid + i * 256;
        float t = (c < D) ? x[rowoff + c] : 0.f;
        v[i] = t;
        ss += t * t;
    }
    red[tid] = ss;
    __syncthreads();
#pragma unroll
    for (int o = 128; o > 0; o >>= 1) {
        if (tid < o) red[tid] += red[tid + o];
        __syncthreads();
    }
    float rms = rsqrtf(red[0] / (float)D + FF_EPS);
    __syncthreads();
#pragma unroll
    for (int i = 0; i < 32; ++i) v[i] *= rms;

    {
        float m = 0.f;
#pragma unroll
        for (int i = 0; i < 32; ++i) {
            int c = tid + i * 256;
            if (c < D) m = fmaxf(m, fabsf(g1[c] * v[i]));
        }
        red[tid] = m;
        __syncthreads();
#pragma unroll
        for (int o = 128; o > 0; o >>= 1) {
            if (tid < o) red[tid] = fmaxf(red[tid], red[tid + o]);
            __syncthreads();
        }
        float scale = 127.f / fmaxf(red[0], 1e-4f);
        if (tid == 0) deq1[blockIdx.x] = 1.f / scale;
        __syncthreads();
#pragma unroll
        for (int i = 0; i < 32; ++i) {
            int c = tid + i * 256;
            if (c < D) {
                float qv = fminf(fmaxf(rintf(g1[c] * v[i] * scale), -128.f), 127.f);
                q1[rowoff + c] = (signed char)(int)qv;
            }
        }
    }
    if (q2) {
        __syncthreads();
        float m = 0.f;
#pragma unroll
        for (int i = 0; i < 32; ++i) {
            int c = tid + i * 256;
            if (c < D) m = fmaxf(m, fabsf(g2[c] * v[i]));
        }
        red[tid] = m;
        __syncthreads();
#pragma unroll
        for (int o = 128; o > 0; o >>= 1) {
            if (tid < o) red[tid] = fmaxf(red[tid], red[tid + o]);
            __syncthreads();
        }
        float scale = 127.f / fmaxf(red[0], 1e-4f);
        if (tid == 0) deq2[blockIdx.x] = 1.f / scale;
        __syncthreads();
#pragma unroll
        for (int i = 0; i < 32; ++i) {
            int c = tid + i * 256;
            if (c < D) {
                float qv = fminf(fmaxf(rintf(g2[c] * v[i] * scale), -128.f), 127.f);
                q2[rowoff + c] = (signed char)(int)qv;
            }
        }
    }
}

__device__ __forceinline__ v8i load_a_frag(const signed char* p) {
    int4 r0 = *(const int4*)(p);
    int4 r1 = *(const int4*)(p + 32);
    v8i a = {r0.x, r0.y, r0.z, r0.w, r1.x, r1.y, r1.z, r1.w};
    return a;
}
__device__ __forceinline__ v8i load_b_frag(const signed char* p) {
    int4 r0 = *(const int4*)(p);
    int4 r1 = *(const int4*)(p + 32);
    v8i b = {r0.x, r0.y, r0.z, r0.w, r1.x, r1.y, r1.z, r1.w};
    return b;
}

__global__ void __launch_bounds__(256, 1)
gemm_up(const signed char* __restrict__ xq1,
        const signed char* __restrict__ xq2,
        const signed char* __restrict__ wq1,
        const signed char* __restrict__ wq2,
        const float* __restrict__ deq1, const float* __restrict__ deq2,
        const float* __restrict__ scl, float* __restrict__ h, int K, int N) {
    const int lane = threadIdx.x & 31;
    const int wave = threadIdx.x >> 5;
    const int wm = wave >> 1, wn = wave & 1;
    const int half = lane >> 4, l16 = lane & 15;
    const int m0 = blockIdx.x * 128 + wm * 32;
    const int n0 = blockIdx.y * 64 + wn * 32;

    v8i acc1[2][2] = {};
    v8i acc2[2][2] = {};

#pragma unroll 1
    for (int kc = 0; kc < K; kc += 64) {
        v8i a1[2], a2[2], b1[2], b2[2];
#pragma unroll
        for (int mi = 0; mi < 2; ++mi) {
            size_t ro = (size_t)(m0 + mi * 16 + l16) * K + kc + half * 16;
            a1[mi] = load_a_frag(xq1 + ro);
            a2[mi] = load_a_frag(xq2 + ro);
            __builtin_prefetch(xq1 + ro + 64, 0, 3);
            __builtin_prefetch(xq2 + ro + 64, 0, 3);
        }
#pragma unroll
        for (int ni = 0; ni < 2; ++ni) {
            size_t co = (size_t)(n0 + ni * 16 + l16) * K + kc + half * 16;
            b1[ni] = load_b_frag(wq1 + co);
            b2[ni] = load_b_frag(wq2 + co);
        }
#pragma unroll
        for (int mi = 0; mi < 2; ++mi)
#pragma unroll
            for (int ni = 0; ni < 2; ++ni) {
                acc1[mi][ni] = __builtin_amdgcn_wmma_i32_16x16x64_iu8(
                    true, a1[mi], true, b1[ni], acc1[mi][ni], false, false);
                acc2[mi][ni] = __builtin_amdgcn_wmma_i32_16x16x64_iu8(
                    true, a2[mi], true, b2[ni], acc2[mi][ni], false, false);
            }
    }

    const float t1 = scl[8], t2 = scl[9];
#pragma unroll
    for (int mi = 0; mi < 2; ++mi) {
#pragma unroll
        for (int r = 0; r < 8; ++r) {
            int grow = m0 + mi * 16 + half * 8 + r;
            float f1 = t1 * deq1[grow];
            float f2 = t2 * deq2[grow];
            size_t base = (size_t)grow * N;
#pragma unroll
            for (int ni = 0; ni < 2; ++ni) {
                float u1 = (float)acc1[mi][ni][r] * f1;
                float u2 = (float)acc2[mi][ni][r] * f2;
                float sw = u1 / (1.f + __expf(-u1));
                h[base + n0 + ni * 16 + l16] = sw * u2;
            }
        }
    }
}

__global__ void __launch_bounds__(256, 1)
gemm_down(const signed char* __restrict__ hq,
          const signed char* __restrict__ wq3,
          const float* __restrict__ deqh, const float* __restrict__ scl,
          float* __restrict__ out, int K, int N) {
    const int lane = threadIdx.x & 31;
    const int wave = threadIdx.x >> 5;
    const int wm = wave & 1, wn = wave >> 1;
    const int half = lane >> 4, l16 = lane & 15;
    const int m0 = blockIdx.x * 128 + wm * 64;
    const int n0 = blockIdx.y * 128 + wn * 32;

    v8i acc[4][2] = {};

#pragma unroll 1
    for (int kc = 0; kc < K; kc += 64) {
        v8i a[4], b[2];
#pragma unroll
        for (int mi = 0; mi < 4; ++mi) {
            size_t ro = (size_t)(m0 + mi * 16 + l16) * K + kc + half * 16;
            a[mi] = load_a_frag(hq + ro);
            __builtin_prefetch(hq + ro + 64, 0, 3);
        }
#pragma unroll
        for (int ni = 0; ni < 2; ++ni) {
            size_t co = (size_t)(n0 + ni * 16 + l16) * K + kc + half * 16;
            b[ni] = load_b_frag(wq3 + co);
        }
#pragma unroll
        for (int mi = 0; mi < 4; ++mi)
#pragma unroll
            for (int ni = 0; ni < 2; ++ni)
                acc[mi][ni] = __builtin_amdgcn_wmma_i32_16x16x64_iu8(
                    true, a[mi], true, b[ni], acc[mi][ni], false, false);
    }

    const float t3 = scl[10];
#pragma unroll
    for (int mi = 0; mi < 4; ++mi) {
#pragma unroll
        for (int r = 0; r < 8; ++r) {
            int grow = m0 + mi * 16 + half * 8 + r;
            float f = t3 * deqh[grow];
            size_t base = (size_t)grow * N;
#pragma unroll
            for (int ni = 0; ni < 2; ++ni)
                out[base + n0 + ni * 16 + l16] = (float)acc[mi][ni][r] * f;
        }
    }
}

extern "C" void kernel_launch(void* const* d_in, const int* in_sizes, int n_in,
                              void* d_out, int out_size, void* d_ws,
                              size_t ws_size, hipStream_t stream) {
    const float* x  = (const float*)d_in[0];
    const float* w1 = (const float*)d_in[1];
    const float* g1 = (const float*)d_in[2];
    const float* w2 = (const float*)d_in[3];
    const float* g2 = (const float*)d_in[4];
    const float* w3 = (const float*)d_in[5];
    const float* g3 = (const float*)d_in[6];

    const int D = in_sizes[2];
    const int H = in_sizes[6];
    const int M = in_sizes[0] / D;
    const size_t WN = (size_t)H * (size_t)D;

    char* ws = (char*)d_ws;
    size_t off = 0;
    float* scl = (float*)(ws + off); off += 256;
    float* part = (float*)(ws + off); off += 3 * 1024 * 4;
    off = (off + 255) & ~(size_t)255;
    signed char* xq1 = (signed char*)(ws + off); off += (size_t)M * D;
    signed char* xq2 = (signed char*)(ws + off); off += (size_t)M * D;
    signed char* wq1 = (signed char*)(ws + off); off += WN;
    signed char* wq2 = (signed char*)(ws + off); off += WN;
    signed char* wq3 = (signed char*)(ws + off); off += WN;
    float* deq1 = (float*)(ws + off); off += (size_t)M * 4;
    float* deq2 = (float*)(ws + off); off += (size_t)M * 4;
    float* deqh = (float*)(ws + off); off += (size_t)M * 4;
    signed char* hq = (signed char*)(ws + off); off += (size_t)M * H;
    off = (off + 255) & ~(size_t)255;
    float* hbuf = (float*)(ws + off); off += (size_t)M * H * 4;

    absmean_partial<<<1024, 256, 0, stream>>>(w1, WN, part + 0);
    absmean_partial<<<1024, 256, 0, stream>>>(w2, WN, part + 1024);
    absmean_partial<<<1024, 256, 0, stream>>>(w3, WN, part + 2048);
    finalize_scales<<<1, 256, 0, stream>>>(part, scl, 1.0f / (float)WN);

    ternary_quant_w<<<2048, 256, 0, stream>>>(w1, scl, 0, wq1, WN);
    ternary_quant_w<<<2048, 256, 0, stream>>>(w2, scl, 1, wq2, WN);
    ternary_quant_w<<<2048, 256, 0, stream>>>(w3, scl, 2, wq3, WN);

    rmsnorm_actquant<<<M, 256, 0, stream>>>(x, D, g1, xq1, deq1, g2, xq2, deq2);

    dim3 gu(M / 128, H / 64);
    gemm_up<<<gu, 256, 0, stream>>>(xq1, xq2, wq1, wq2, deq1, deq2, scl, hbuf,
                                    D, H);

    rmsnorm_actquant<<<M, 256, 0, stream>>>(hbuf, H, g3, hq, deqh, nullptr,
                                            nullptr, nullptr);

    dim3 gd(M / 128, D / 128);
    gemm_down<<<gd, 256, 0, stream>>>(hq, wq3, deqh, scl, (float*)d_out, H, D);
}
